// TurboQuantWrapper_81192061763791
// MI455X (gfx1250) — matrix operand order repaired and checked by reading, not compiled, not run on hardware
//
#include <hip/hip_runtime.h>
#include <hip/hip_bf16.h>
#include <stdint.h>

// ---------------------------------------------------------------------------
// Problem constants (from the reference)
// ---------------------------------------------------------------------------
#define IN_DIM     4096
#define OUT_DIM    4096
#define GROUP_SIZE 128
#define N_GROUPS   (IN_DIM / GROUP_SIZE)   // 32
#define INV_SQRT_G 0.08838834764831845f    // 1/sqrt(128)

typedef __attribute__((ext_vector_type(16))) __bf16 v16bf;
typedef __attribute__((ext_vector_type(8)))  __bf16 v8bf;
typedef __attribute__((ext_vector_type(8)))  float  v8f;
typedef __attribute__((ext_vector_type(4)))  int    v4i;

// ---------------------------------------------------------------------------
// CDNA5 async global->LDS path (guarded; fallback = reg + ds_store)
// ---------------------------------------------------------------------------
#if defined(__has_builtin)
#if __has_builtin(__builtin_amdgcn_global_load_async_to_lds_b128)
#define USE_ASYNC_LDS 1
#endif
#endif

#ifdef USE_ASYNC_LDS
__device__ __forceinline__ void async_copy_b128(const void* g, void* lds)
{
    // Builtin prototype (clang-22): (AS1 v4i* src, AS3 v4i* dst, imm off, imm cpol).
    // generic->AS1 is an identity value cast; generic LDS addr low 32 bits
    // are the LDS byte offset (ISA 10.2).
    __builtin_amdgcn_global_load_async_to_lds_b128(
        (__attribute__((address_space(1))) v4i*)(uintptr_t)g,
        (__attribute__((address_space(3))) v4i*)(uint32_t)(uintptr_t)lds,
        0, 0);
}
__device__ __forceinline__ void wait_asynccnt0()
{
#if __has_builtin(__builtin_amdgcn_s_wait_asynccnt)
    __builtin_amdgcn_s_wait_asynccnt(0);
#else
    asm volatile("s_wait_asynccnt 0x0" ::: "memory");
#endif
}
#endif

// ---------------------------------------------------------------------------
// Kernel 1: decode quantized weights -> bf16 w[o][k], k contiguous.
// One 128-thread half-block per weight group (row of packed_weight).
// Faithful to reference wht(): 7 iterations of the stride-64 split butterfly.
// ---------------------------------------------------------------------------
__global__ void decode_wq_kernel(const int* __restrict__ codes,
                                 const float* __restrict__ norms,     // (OUT, N_GROUPS) flat
                                 const float* __restrict__ signs1,    // (128)
                                 const float* __restrict__ signs2,    // (128)
                                 const float* __restrict__ cent,      // (8)
                                 __bf16* __restrict__ w)              // (OUT, IN) bf16
{
    __shared__ float buf[2][GROUP_SIZE];

    const int tid  = threadIdx.x;
    const int half = tid >> 7;
    const int i    = tid & 127;
    const int r    = blockIdx.x * 2 + half;       // packed row, 0..131071
    const int o    = r >> 5;
    const int gi   = r & 31;

    const int   code = codes[(size_t)r * GROUP_SIZE + i];
    const float u    = cent[code] * norms[r] * signs2[i];

    buf[half][i] = u;
    __syncthreads();

    float v = u;
#pragma unroll
    for (int it = 0; it < 7; ++it) {
        const float lo = buf[half][i & 63];
        const float hi = buf[half][(i & 63) + 64];
        v = (i < 64) ? (lo + hi) : (lo - hi);
        __syncthreads();
        buf[half][i] = v;
        __syncthreads();
    }

    const float wv = v * INV_SQRT_G * signs1[i];
    w[(size_t)o * IN_DIM + gi * GROUP_SIZE + i] = (__bf16)wv;
}

// ---------------------------------------------------------------------------
// Fragment loaders (layouts per CDNA5 ISA 7.12.2, wave32):
//  A 16x32 bf16 : lanes 0-15 -> M=0..15, K {0..7,16..23}; lanes 16-31 -> K {8..15,24..31}
//  B 32x16 bf16 : lanes 0-15 -> N=0..15, K {0..7,16..23}; lanes 16-31 -> K {8..15,24..31}
// LDS holds A as [m][64] (k contig) and B as [n][64] (k contig).
// ---------------------------------------------------------------------------
__device__ __forceinline__ v16bf load_frag_a(const __bf16* base, int m, int ks, int lane)
{
    const int khalf = (lane >> 4) * 8;
    const int row   = lane & 15;
    const __bf16* p = base + (m + row) * 64 + ks + khalf;
    v8bf lo = *(const v8bf*)(p);            // K ks+khalf+0..7
    v8bf hi = *(const v8bf*)(p + 16);       // K ks+khalf+16..23
    return __builtin_shufflevector(lo, hi, 0,1,2,3,4,5,6,7,8,9,10,11,12,13,14,15);
}

__device__ __forceinline__ v16bf load_frag_b(const __bf16* base, int n, int ks, int lane)
{
    const int khalf = (lane >> 4) * 8;
    const int col   = lane & 15;
    const __bf16* p = base + (n + col) * 64 + ks + khalf;
    v8bf lo = *(const v8bf*)(p);            // K ks+khalf+0..7
    v8bf hi = *(const v8bf*)(p + 16);       // K ks+khalf+16..23
    return __builtin_shufflevector(lo, hi, 0,1,2,3,4,5,6,7,8,9,10,11,12,13,14,15);
}

#define TILE_ELEMS (128 * 64)
#define NITER      (IN_DIM / 64)

// ---------------------------------------------------------------------------
// Kernel 2: out[M,4096] = x[M,4096](fp32->bf16) * w^T + bias, bf16 WMMA.
// Block: 256 threads (8 waves), tile 128(M) x 128(N), K step 64.
// Double-buffered LDS, register-prefetched A, async global->LDS B,
// single barrier per K iteration.
// ---------------------------------------------------------------------------
__global__ void __launch_bounds__(256)
gemm_bf16_wmma_kernel(const float* __restrict__ x,
                      const __bf16* __restrict__ w,      // (OUT, IN) bf16
                      const float* __restrict__ bias,    // (OUT)
                      float* __restrict__ out,           // (M, OUT)
                      int M)
{
    __shared__ __align__(16) __bf16 As[2 * TILE_ELEMS];
    __shared__ __align__(16) __bf16 Bs[2 * TILE_ELEMS];

    const int tid   = threadIdx.x;
    const int lane  = tid & 31;
    const int wave  = tid >> 5;
    const int m_off = (wave & 3) * 32;
    const int n_off = (wave >> 2) * 64;

    const int nBase = blockIdx.x * 128;  // N fastest -> x panel reuse in L2
    const int mBase = blockIdx.y * 128;

    v8f acc[2][4] = {};

    // staging roles: thread t handles row (t>>1), 32-element k-segment (t&1)
    const int srow = tid >> 1;
    const int sseg = (tid & 1) * 32;

    const float*  xsrc = x + (size_t)(mBase + srow) * IN_DIM + sseg;
    const __bf16* wsrc = w + (size_t)(nBase + srow) * IN_DIM + sseg;
    const int lds_off = srow * 64 + sseg;

    // ---- prologue: A tile 0 -> regs, B tile 0 -> Bs[0] ----
    float4 areg[8];
#pragma unroll
    for (int j = 0; j < 8; ++j)
        areg[j] = ((const float4*)xsrc)[j];

#ifdef USE_ASYNC_LDS
#pragma unroll
    for (int j = 0; j < 4; ++j)
        async_copy_b128(wsrc + j * 8, Bs + lds_off + j * 8);
#else
    float4 breg[4];
#pragma unroll
    for (int j = 0; j < 4; ++j)
        breg[j] = ((const float4*)wsrc)[j];
#endif

    for (int it = 0; it < NITER; ++it) {
        const int p = it & 1;
        __bf16* Ap = As + p * TILE_ELEMS;
        __bf16* Bp = Bs + p * TILE_ELEMS;
        __bf16* Bq = Bs + (p ^ 1) * TILE_ELEMS;
        const int knext = (it + 1) * 64;

        // ---- convert + store A regs -> Ap ----
        {
            __bf16* adst = Ap + lds_off;
#pragma unroll
            for (int c = 0; c < 4; ++c) {
                v8bf pk;
                pk[0] = (__bf16)areg[2*c].x;   pk[1] = (__bf16)areg[2*c].y;
                pk[2] = (__bf16)areg[2*c].z;   pk[3] = (__bf16)areg[2*c].w;
                pk[4] = (__bf16)areg[2*c+1].x; pk[5] = (__bf16)areg[2*c+1].y;
                pk[6] = (__bf16)areg[2*c+1].z; pk[7] = (__bf16)areg[2*c+1].w;
                ((v8bf*)adst)[c] = pk;
            }
        }
#ifndef USE_ASYNC_LDS
        {
            __bf16* bdst = Bp + lds_off;
#pragma unroll
            for (int j = 0; j < 4; ++j)
                ((float4*)bdst)[j] = breg[j];
        }
#endif

        // ---- prefetch next A tile into regs ----
        if (it + 1 < NITER) {
#pragma unroll
            for (int j = 0; j < 8; ++j)
                areg[j] = ((const float4*)(xsrc + knext))[j];
        }

#ifdef USE_ASYNC_LDS
        wait_asynccnt0();          // B tile `it` fully landed in Bp
#endif
        __syncthreads();           // Ap/Bp staged; old reads of Bq/Aq done

        // ---- issue next B tile into the other buffer ----
        if (it + 1 < NITER) {
#ifdef USE_ASYNC_LDS
#pragma unroll
            for (int j = 0; j < 4; ++j)
                async_copy_b128(wsrc + knext + j * 8, Bq + lds_off + j * 8);
#else
#pragma unroll
            for (int j = 0; j < 4; ++j)
                breg[j] = ((const float4*)(wsrc + knext))[j];
#endif
        }

        // ---- compute: 2 k-steps x (2 M x 4 N) = 16 WMMA ----
#pragma unroll
        for (int ks = 0; ks < 64; ks += 32) {
            v16bf af[2], bfr[4];
#pragma unroll
            for (int mm = 0; mm < 2; ++mm)
                af[mm] = load_frag_a(Ap, m_off + mm * 16, ks, lane);
#pragma unroll
            for (int nn = 0; nn < 4; ++nn)
                bfr[nn] = load_frag_b(Bp, n_off + nn * 16, ks, lane);
#pragma unroll
            for (int mm = 0; mm < 2; ++mm)
#pragma unroll
                for (int nn = 0; nn < 4; ++nn)
                    acc[mm][nn] = __builtin_amdgcn_wmma_f32_16x16x32_bf16(
                        false, af[mm], false, bfr[nn], (short)0, acc[mm][nn],
                        false, false);
        }
    }

    // ---- epilogue: C layout = VGPR r -> M = r + 8*(lane>=16), N = lane&15 ----
#pragma unroll
    for (int mm = 0; mm < 2; ++mm) {
#pragma unroll
        for (int nn = 0; nn < 4; ++nn) {
            const int col = nBase + n_off + nn * 16 + (lane & 15);
            const float bv = bias[col];
            const int rowb = mBase + m_off + mm * 16 + (lane >> 4) * 8;
#pragma unroll
            for (int r = 0; r < 8; ++r)
                out[(size_t)(rowb + r) * OUT_DIM + col] = acc[mm][nn][r] + bv;
        }
    }
}

// ---------------------------------------------------------------------------
// Launcher
// ---------------------------------------------------------------------------
extern "C" void kernel_launch(void* const* d_in, const int* in_sizes, int n_in,
                              void* d_out, int out_size, void* d_ws, size_t ws_size,
                              hipStream_t stream)
{
    const float* x      = (const float*)d_in[0];
    const int*   codes  = (const int*)  d_in[1];
    const float* norms  = (const float*)d_in[2];
    const float* signs1 = (const float*)d_in[3];
    const float* signs2 = (const float*)d_in[4];
    const float* cent   = (const float*)d_in[5];
    const float* bias   = (const float*)d_in[6];
    float*  out  = (float*)d_out;
    __bf16* wdec = (__bf16*)d_ws;           // 4096*4096*2 = 33.5 MB of scratch

    const int M = in_sizes[0] / IN_DIM;     // 16384

    decode_wq_kernel<<<(OUT_DIM * N_GROUPS) / 2, 256, 0, stream>>>(
        codes, norms, signs1, signs2, cent, wdec);

    dim3 grid(OUT_DIM / 128, M / 128);
    gemm_bf16_wmma_kernel<<<grid, 256, 0, stream>>>(x, wdec, bias, out, M);
}
